// InterpretableMultiHeadAttention_88038239633915
// MI455X (gfx1250) — matrix operand order repaired and checked by reading, not compiled, not run on hardware
//
#include <hip/hip_runtime.h>

// Problem constants (match reference)
#define S_LEN   1024
#define D_MODEL 512
#define NHEAD   8
#define DK      64
#define BATCH   8

typedef __attribute__((ext_vector_type(16))) __bf16       v16bf;
typedef __attribute__((ext_vector_type(2)))  __bf16       v2bf;
typedef __attribute__((ext_vector_type(8)))  float        v8f;
typedef __attribute__((ext_vector_type(4)))  float        f4;
typedef __attribute__((ext_vector_type(4)))  unsigned int u4;
typedef __attribute__((ext_vector_type(2)))  unsigned int u2;

union BFrag {
    v16bf v;
    unsigned short u[16];
    u4 q[2];
};

#if defined(__has_builtin)
#if __has_builtin(__builtin_amdgcn_cvt_pk_bf16_f32)
#define HAVE_PK_BF16 1
#endif
#endif

__device__ __forceinline__ unsigned short f32_to_bf16(float f) {
    unsigned int x = __float_as_uint(f);
    x += 0x7FFFu + ((x >> 16) & 1u);   // round-to-nearest-even
    return (unsigned short)(x >> 16);
}

// 2x fp32 -> packed 2x bf16 (RNE). Uses v_cvt_pk_bf16_f32 when available.
__device__ __forceinline__ void cvt2(float a, float b, unsigned short* d) {
#ifdef HAVE_PK_BF16
    union { v2bf v; unsigned short u[2]; } t;
    t.v = __builtin_amdgcn_cvt_pk_bf16_f32(a, b);
    d[0] = t.u[0]; d[1] = t.u[1];
#else
    d[0] = f32_to_bf16(a); d[1] = f32_to_bf16(b);
#endif
}

__device__ __forceinline__ void cvt4(f4 x, unsigned short* d) {
    cvt2(x.x, x.y, d);
    cvt2(x.z, x.w, d + 2);
}

// A-matrix fragment (16x32 bf16, MxK): lane&15 = M row; lane>>4 selects K bank.
// elements 0..7 -> K = kb..kb+7, elements 8..15 -> K = kb+16..kb+23 (kb = k0 + hi*8)
__device__ __forceinline__ void load_a_f32(const float* row, int kb, BFrag& f) {
    cvt4(*(const f4*)(row + kb),      &f.u[0]);
    cvt4(*(const f4*)(row + kb + 4),  &f.u[4]);
    cvt4(*(const f4*)(row + kb + 16), &f.u[8]);
    cvt4(*(const f4*)(row + kb + 20), &f.u[12]);
}
__device__ __forceinline__ void load_a_bf16(const unsigned short* row, int kb, BFrag& f) {
    f.q[0] = *(const u4*)(row + kb);        // K = kb .. kb+7
    f.q[1] = *(const u4*)(row + kb + 16);   // K = kb+16 .. kb+23
}
// B-matrix fragment (32x16 bf16, KxN): lane&15 = N col; kb2 = k0 + hi*16;
// elements 0..7 -> K = k0+hi*8..+7, elements 8..15 -> K = k0+16+hi*8..+7 (same as A)
__device__ __forceinline__ void load_b_bf16(const unsigned short* row, int kb2, BFrag& f) {
    const int kb = kb2 - 8 * ((kb2 >> 4) & 1);
    f.q[0] = *(const u4*)(row + kb);
    f.q[1] = *(const u4*)(row + kb + 16);
}

__device__ __forceinline__ v8f wmma_bf16(const BFrag& a, const BFrag& b, v8f c) {
    return __builtin_amdgcn_wmma_f32_16x16x32_bf16(false, a.v, false, b.v,
                                                   (short)0, c, false, false);
}

// ---------------------------------------------------------------------------
// Kernel 0: streaming fp32 -> bf16 conversion (convert-once for all GEMM
// operands; removes per-tile conversion VALU from the projection kernel).
// ---------------------------------------------------------------------------
__global__ __launch_bounds__(256) void cvt_kernel(
    const float* __restrict__ s, unsigned short* __restrict__ d, int n4)
{
    int i = blockIdx.x * blockDim.x + threadIdx.x;
    const int stride = gridDim.x * blockDim.x;
    for (; i < n4; i += stride) {
        f4 x = *(const f4*)(s + 4 * (size_t)i);
        union { unsigned short u[4]; u2 w; } t;
        cvt4(x, t.u);
        *(u2*)(d + 4 * (size_t)i) = t.w;
    }
}

// ---------------------------------------------------------------------------
// Kernel 1: projections (pure bf16 loads -> WMMA).
// 136 GEMMs C[1024x64] = A[1024x512] * W[64x512]^T
//   job  0..63 : qh[b,h]  (scaled by 1/sqrt(64)=0.125, bf16 row-major [S][64])
//   job 64..127: kh[b,h]  (bf16 row-major [S][64])
//   job128..135: vs[b]    (bf16 stored TRANSPOSED [64][S] for attn*V B-frags)
// grid = 136*32 blocks of 256 threads; each wave -> one 16x16 C tile.
// ---------------------------------------------------------------------------
__global__ __launch_bounds__(256) void proj_kernel(
    const unsigned short* __restrict__ qb, const unsigned short* __restrict__ kb_,
    const unsigned short* __restrict__ vb,
    const unsigned short* __restrict__ Wqb, const unsigned short* __restrict__ Wkb,
    const unsigned short* __restrict__ Wvb,
    unsigned short* __restrict__ qh, unsigned short* __restrict__ kh,
    unsigned short* __restrict__ vsT)
{
    const int job    = blockIdx.x >> 5;
    const int rowblk = blockIdx.x & 31;
    const int w    = threadIdx.x >> 5;
    const int lane = threadIdx.x & 31;
    const int l15  = lane & 15;
    const int hi   = lane >> 4;

    const unsigned short* A; const unsigned short* W;
    unsigned short* dst; int transposed = 0; float scale = 1.0f;
    if (job < 64) {
        int b = job >> 3, h = job & 7;
        A = qb + (size_t)b * S_LEN * D_MODEL;
        W = Wqb + (size_t)h * DK * D_MODEL;
        dst = qh + (size_t)(b * NHEAD + h) * S_LEN * DK;
        scale = 0.125f;                      // fold 1/sqrt(d_k)
    } else if (job < 128) {
        int j = job - 64, b = j >> 3, h = j & 7;
        A = kb_ + (size_t)b * S_LEN * D_MODEL;
        W = Wkb + (size_t)h * DK * D_MODEL;
        dst = kh + (size_t)(b * NHEAD + h) * S_LEN * DK;
    } else {
        int b = job - 128;
        A = vb + (size_t)b * S_LEN * D_MODEL;
        W = Wvb;
        dst = vsT + (size_t)b * DK * S_LEN;
        transposed = 1;
    }

    const int m0 = rowblk * 32 + (w & 1) * 16;
    const int n0 = (w >> 1) * 16;
    const unsigned short* arow = A + (size_t)(m0 + l15) * D_MODEL;
    const unsigned short* wrow = W + (size_t)(n0 + l15) * D_MODEL; // W^T col = W row

    v8f c = {};
    for (int k0 = 0; k0 < D_MODEL; k0 += 32) {
        BFrag af, bf;
        load_a_bf16(arow, k0 + hi * 8, af);
        load_b_bf16(wrow, k0 + hi * 16, bf);
        c = wmma_bf16(af, bf, c);
    }

    if (!transposed) {
        const int col = n0 + l15;
        #pragma unroll
        for (int r = 0; r < 8; ++r) {
            int row = m0 + hi * 8 + r;
            dst[(size_t)row * DK + col] = f32_to_bf16(c[r] * scale);
        }
    } else {
        const int e  = n0 + l15;
        const int s0 = m0 + hi * 8;
        union { unsigned short u[8]; u4 q; } pk;
        #pragma unroll
        for (int r = 0; r < 8; r += 2) cvt2(c[r], c[r + 1], &pk.u[r]);
        *(u4*)(dst + (size_t)e * S_LEN + s0) = pk.q;    // 8 consecutive s -> 16B store
    }
}

// ---------------------------------------------------------------------------
// Kernel 2: per (b, h, 32-query block): scores -> softmax -> attn out (NT
// stores) -> heads = attn * vs accumulated across heads via f32 atomics.
// 8 waves / 256 threads; 32x1028 fp32 score tile in dynamic LDS (~130KB,
// CDNA5 WGP has 320KB).
// ---------------------------------------------------------------------------
#define SROW 1028   // 1024 + pad: keeps 16B alignment, shifts banks by 4 per row

__global__ __launch_bounds__(256) void attn_kernel(
    const unsigned short* __restrict__ qh, const unsigned short* __restrict__ kh,
    const unsigned short* __restrict__ vsT,
    float* __restrict__ attn_out, float* __restrict__ headsum)
{
    extern __shared__ float smem[];
    float* sc     = smem;                    // [32][SROW]
    float* red    = smem + 32 * SROW;        // [32*8] partial reductions
    float* rowmax = red + 256;               // [32]
    float* rowsum = rowmax + 32;             // [32]

    const int qblk = blockIdx.x & 31;
    const int bh   = blockIdx.x >> 5;
    const int b    = bh >> 3, h = bh & 7;

    const int w    = threadIdx.x >> 5;
    const int lane = threadIdx.x & 31;
    const int l15  = lane & 15;
    const int hi   = lane >> 4;

    const unsigned short* qh_bh = qh + (size_t)(b * NHEAD + h) * S_LEN * DK;
    const unsigned short* kh_bh = kh + (size_t)(b * NHEAD + h) * S_LEN * DK;

    // ---- Phase 1: scores[32][1024] = (qh/sqrt(dk)) * kh^T ----
    {
        const int qt = w & 1;          // which 16-row query tile
        const int ks = w >> 1;         // key slice: 16 k-tiles each
        const unsigned short* ar = qh_bh + (size_t)(qblk * 32 + qt * 16 + l15) * DK;
        BFrag a0, a1;
        load_a_bf16(ar, 0 + hi * 8, a0);    // e in [0,32)
        load_a_bf16(ar, 32 + hi * 8, a1);   // e in [32,64)
        for (int t = 0; t < 16; ++t) {
            const int n0 = (ks * 16 + t) * 16;
            const unsigned short* br = kh_bh + (size_t)(n0 + l15) * DK;
            __builtin_prefetch((const void*)(br + 16 * DK), 0, 0); // -> global_prefetch
            BFrag b0, b1;
            load_b_bf16(br, 0 + hi * 16, b0);
            load_b_bf16(br, 32 + hi * 16, b1);
            v8f c = {};
            c = wmma_bf16(a0, b0, c);
            c = wmma_bf16(a1, b1, c);
            float* dstp = sc + (size_t)(qt * 16 + hi * 8) * SROW + n0 + l15;
            #pragma unroll
            for (int r = 0; r < 8; ++r) dstp[(size_t)r * SROW] = c[r];
        }
    }
    __syncthreads();

    // ---- Phase 2: softmax over 1024 keys; stream attn to global (NT) ----
    {
        const int row = threadIdx.x >> 3;   // 0..31
        const int sub = threadIdx.x & 7;    // 8 threads/row, 128 cols each
        float* srow = sc + (size_t)row * SROW + sub * 128;

        float m = -3.4e38f;
        for (int i = 0; i < 128; i += 4) {
            f4 x = *(const f4*)(srow + i);
            m = fmaxf(m, fmaxf(fmaxf(x.x, x.y), fmaxf(x.z, x.w)));
        }
        red[row * 8 + sub] = m;
        __syncthreads();
        if (sub == 0) {
            float mm = red[row * 8];
            #pragma unroll
            for (int i = 1; i < 8; ++i) mm = fmaxf(mm, red[row * 8 + i]);
            rowmax[row] = mm;
        }
        __syncthreads();
        const float mm = rowmax[row];
        float s = 0.f;
        for (int i = 0; i < 128; i += 4) {
            f4 x = *(const f4*)(srow + i);
            x.x = __expf(x.x - mm); x.y = __expf(x.y - mm);
            x.z = __expf(x.z - mm); x.w = __expf(x.w - mm);
            *(f4*)(srow + i) = x;
            s += x.x + x.y + x.z + x.w;
        }
        red[row * 8 + sub] = s;
        __syncthreads();
        if (sub == 0) {
            float ss = 0.f;
            #pragma unroll
            for (int i = 0; i < 8; ++i) ss += red[row * 8 + i];
            rowsum[row] = ss;
        }
        __syncthreads();
        const float inv = 1.0f / rowsum[row];
        // attn layout [H][B][S][S]
        float* gout = attn_out +
            ((((size_t)h * BATCH + b) * S_LEN) + (size_t)(qblk * 32 + row)) * S_LEN +
            sub * 128;
        for (int i = 0; i < 128; i += 4) {
            f4 x = *(const f4*)(srow + i);
            x.x *= inv; x.y *= inv; x.z *= inv; x.w *= inv;
            *(f4*)(srow + i) = x;                       // keep for heads GEMM
            __builtin_nontemporal_store(x, (f4*)(gout + i));  // 268MB stream: bypass caches
        }
    }
    __syncthreads();

    // ---- Phase 3: heads[32][64] = attn[32][1024] * vs[1024][64], sum over h ----
    {
        const int qt = w & 1;
        const int ct = w >> 1;          // 4 col tiles of 16 -> 64
        const int n0 = ct * 16;
        const unsigned short* vrow = vsT + ((size_t)b * DK + n0 + l15) * S_LEN;
        const float* arow = sc + (size_t)(qt * 16 + l15) * SROW;
        v8f c = {};
        for (int k0 = 0; k0 < S_LEN; k0 += 32) {
            BFrag af, bf;
            load_a_f32(arow, k0 + hi * 8, af);      // LDS f32 -> bf16 (v_cvt_pk_bf16_f32)
            load_b_bf16(vrow, k0 + hi * 16, bf);
            c = wmma_bf16(af, bf, c);
        }
        #pragma unroll
        for (int r = 0; r < 8; ++r) {
            int qrow = qblk * 32 + qt * 16 + hi * 8 + r;
            atomicAdd(&headsum[((size_t)b * S_LEN + qrow) * DK + n0 + l15], c[r]);
        }
    }
}

// ---------------------------------------------------------------------------
// Kernel 3: out[b][s][d] = (headsum[b][s][:]/8) . Wo[d][:]
// GEMM M=1024, N=512, K=64 per batch. grid = 8*32*8 blocks.
// ---------------------------------------------------------------------------
__global__ __launch_bounds__(256) void out_kernel(
    const float* __restrict__ headsum, const unsigned short* __restrict__ Wob,
    float* __restrict__ out)
{
    const int dblk = blockIdx.x & 7;
    const int qblk = (blockIdx.x >> 3) & 31;
    const int b    = blockIdx.x >> 8;
    const int w    = threadIdx.x >> 5;
    const int lane = threadIdx.x & 31;
    const int l15  = lane & 15;
    const int hi   = lane >> 4;
    const int qt = w & 1, ct = w >> 1;

    const int row0 = qblk * 32 + qt * 16;
    const int col0 = dblk * 64 + ct * 16;
    const float*          ar = headsum + ((size_t)b * S_LEN + row0 + l15) * DK;
    const unsigned short* wr = Wob + (size_t)(col0 + l15) * DK; // Wo[d][e]: contiguous in e

    v8f c = {};
    for (int e0 = 0; e0 < DK; e0 += 32) {
        BFrag af, bf;
        load_a_f32(ar, e0 + hi * 8, af);
        load_b_bf16(wr, e0 + hi * 16, bf);
        c = wmma_bf16(af, bf, c);
    }
    #pragma unroll
    for (int r = 0; r < 8; ++r) {
        int row = row0 + hi * 8 + r;
        out[((size_t)b * S_LEN + row) * D_MODEL + col0 + l15] = c[r] * 0.125f; // mean of 8 heads
    }
}

// ---------------------------------------------------------------------------
extern "C" void kernel_launch(void* const* d_in, const int* in_sizes, int n_in,
                              void* d_out, int out_size, void* d_ws, size_t ws_size,
                              hipStream_t stream)
{
    (void)in_sizes; (void)n_in; (void)out_size; (void)ws_size;
    const float* q  = (const float*)d_in[0];
    const float* k  = (const float*)d_in[1];
    const float* v  = (const float*)d_in[2];
    const float* Wq = (const float*)d_in[3];
    const float* Wk = (const float*)d_in[4];
    const float* Wv = (const float*)d_in[5];
    const float* Wo = (const float*)d_in[6];

    float* out    = (float*)d_out;
    float* attn_o = out + (size_t)BATCH * S_LEN * D_MODEL;   // [H][B][S][S]

    char* ws = (char*)d_ws;
    unsigned short* qh  = (unsigned short*)(ws);              //  8.4 MB bf16
    unsigned short* kh  = (unsigned short*)(ws + 8388608);    //  8.4 MB bf16
    unsigned short* vsT = (unsigned short*)(ws + 16777216);   //  1.0 MB bf16
    float*          hs  = (float*)        (ws + 17825792);    //  2.1 MB f32
    unsigned short* qb  = (unsigned short*)(ws + 19922944);   //  8.4 MB bf16
    unsigned short* kb  = (unsigned short*)(ws + 28311552);   //  8.4 MB bf16
    unsigned short* vb  = (unsigned short*)(ws + 36700160);   //  8.4 MB bf16
    unsigned short* Wqb = (unsigned short*)(ws + 45088768);   //  0.5 MB bf16
    unsigned short* Wkb = (unsigned short*)(ws + 45613056);   //  0.5 MB bf16
    unsigned short* Wvb = (unsigned short*)(ws + 46137344);   //  64 KB bf16
    unsigned short* Wob = (unsigned short*)(ws + 46202880);   //  64 KB bf16

    hipMemsetAsync(hs, 0, (size_t)BATCH * S_LEN * DK * sizeof(float), stream);

    // convert-once: all GEMM operands to bf16
    const int nQKV4 = (BATCH * S_LEN * D_MODEL) / 4;   // 1048576
    cvt_kernel<<<2048, 256, 0, stream>>>(q,  qb,  nQKV4);
    cvt_kernel<<<2048, 256, 0, stream>>>(k,  kb,  nQKV4);
    cvt_kernel<<<2048, 256, 0, stream>>>(v,  vb,  nQKV4);
    cvt_kernel<<<256,  256, 0, stream>>>(Wq, Wqb, (NHEAD * DK * D_MODEL) / 4);
    cvt_kernel<<<256,  256, 0, stream>>>(Wk, Wkb, (NHEAD * DK * D_MODEL) / 4);
    cvt_kernel<<<64,   256, 0, stream>>>(Wv, Wvb, (DK * D_MODEL) / 4);
    cvt_kernel<<<64,   256, 0, stream>>>(Wo, Wob, (D_MODEL * DK) / 4);

    proj_kernel<<<136 * 32, 256, 0, stream>>>(qb, kb, vb, Wqb, Wkb, Wvb, qh, kh, vsT);

    const size_t lds_bytes = (size_t)(32 * SROW + 256 + 32 + 32) * sizeof(float);
    attn_kernel<<<BATCH * NHEAD * (S_LEN / 32), 256, lds_bytes, stream>>>(
        qh, kh, vsT, attn_o, hs);

    out_kernel<<<BATCH * (S_LEN / 32) * (D_MODEL / 64), 256, 0, stream>>>(hs, Wo ? Wob : Wob, out);
}
